// TRXOS_57647051047430
// MI455X (gfx1250) — compile-verified, matrix operand order checked by reading, not run on hardware
//
#include <hip/hip_runtime.h>
#include <hip/hip_bf16.h>

typedef unsigned short u16;
typedef __bf16 bf16;
typedef __attribute__((ext_vector_type(16))) bf16 v16bf;
typedef __attribute__((ext_vector_type(8)))  bf16 v8bf;
typedef __attribute__((ext_vector_type(8)))  float v8f;

#define SEQ_LEN 46
#define D_IN    256
#define D_OUT   128
#define NT      1035              // C(46,2)
#define NSEQ    26                // 25 support + 1 query
#define WAY     5
#define SHOT    5
#define VST     1040              // padded vsT stride (16B aligned rows)

__device__ __forceinline__ u16 f2bf(float f) {
    unsigned u = __float_as_uint(f);
    unsigned r = (u + 0x7FFFu + ((u >> 16) & 1u)) >> 16;   // RNE
    return (u16)r;
}

// Load one bf16 16x32 WMMA fragment (A: M rows / B: N rows stored K-major).
// ISA layout: lanes 0-15 hold row (lane), K chunks [0..7] & [16..23];
// lanes 16-31 hold row (lane-16), K chunks [8..15] & [24..31].
__device__ __forceinline__ v16bf ld_frag(const u16* base, int ld, int row) {
    int lane = threadIdx.x & 31;
    const u16* p = base + row * ld + ((lane & 16) ? 8 : 0);
    v8bf lo = *reinterpret_cast<const v8bf*>(p);
    v8bf hi = *reinterpret_cast<const v8bf*>(p + 16);
    v16bf r;
#pragma unroll
    for (int q = 0; q < 8; ++q) { r[q] = lo[q]; r[q + 8] = hi[q]; }
    return r;
}

// ---------------- tuple indices (lexicographic combinations) ----------------
__global__ void tuple_kernel(int* tupi, int* tupj) {
    int i = threadIdx.x;
    if (i < SEQ_LEN) {
        int off = i * 45 - (i * (i - 1)) / 2;
        for (int j = i + 1; j < SEQ_LEN; ++j) { tupi[off] = i; tupj[off] = j; ++off; }
    }
}

// ---------------- bf16 transposed weights: wT[n][k] = w[k][n] ----------------
__global__ void wprep_kernel(const float* kw, const float* vw, u16* kwT, u16* vwT) {
    for (int idx = blockIdx.x * blockDim.x + threadIdx.x; idx < 512 * 128;
         idx += gridDim.x * blockDim.x) {
        int n = idx / 512, k = idx % 512;
        kwT[idx] = f2bf(kw[k * 128 + n]);
        vwT[idx] = f2bf(vw[k * 128 + n]);
    }
}

__global__ void zero_kernel(float* p, int n) {
    for (int i = blockIdx.x * blockDim.x + threadIdx.x; i < n; i += gridDim.x * blockDim.x)
        p[i] = 0.f;
}

// ---------------- MLP (90->180 relu ->256 relu) + positional encoding -------
__global__ void feat_kernel(const float* ss, const float* qs,
                            const float* w1, const float* b1,
                            const float* w2, const float* b2, u16* featb) {
    int blk = blockIdx.x;                 // 0..NSEQ*46-1
    int seq = blk / SEQ_LEN, frame = blk % SEQ_LEN;
    const float* x = (seq < 25) ? (ss + (seq * SEQ_LEN + frame) * 90)
                                : (qs + frame * 90);
    __shared__ float xs[90];
    __shared__ float hs[180];
    int tid = threadIdx.x;
    if (tid < 90) xs[tid] = x[tid];
    __syncthreads();
    if (tid < 180) {
        float acc = b1[tid];
        for (int k = 0; k < 90; ++k) acc = fmaf(xs[k], w1[k * 180 + tid], acc);
        hs[tid] = fmaxf(acc, 0.f);
    }
    __syncthreads();
    {
        int d = tid;                       // 256 threads == D_IN
        float acc = b2[d];
        for (int k = 0; k < 180; ++k) acc = fmaf(hs[k], w2[k * 256 + d], acc);
        acc = fmaxf(acc, 0.f);
        int kk = d >> 1;
        float div = __expf(-(float)(2 * kk) * (9.210340371976184f / 256.0f));
        float ang = (float)frame * div;
        acc += ((d & 1) ? __cosf(ang) : __sinf(ang)) * 0.1f;
        featb[(seq * SEQ_LEN + frame) * 256 + d] = f2bf(acc);
    }
}

// ---------------- tuple gather + K/V projection (WMMA) + layernorm ----------
__global__ void __launch_bounds__(256)
proj_kernel(const u16* featb, const int* tupi, const int* tupj,
            const u16* kwT, const u16* vwT,
            const float* kb, const float* vb, const float* lng, const float* lnb,
            u16* ks, u16* vsT, float* qv) {
    int tile = blockIdx.x, seq = blockIdx.y;
    __shared__ __align__(16) u16 At[16][512];
    __shared__ float Ck[16][128];
    __shared__ float Cv[16][128];
    __shared__ int fi[16], fj[16], tv[16];
    int tid = threadIdx.x;
    if (tid < 16) {
        int t = tile * 16 + tid;
        int tc = t < NT ? t : NT - 1;
        tv[tid] = t; fi[tid] = tupi[tc]; fj[tid] = tupj[tc];
    }
    __syncthreads();
    // stage A: 16 rows x 512 bf16 (concat frame-i | frame-j features),
    // gathered straight into LDS with CDNA5 async loads (ASYNCcnt-tracked)
    const u16* fb = featb + seq * SEQ_LEN * 256;
    for (int c = tid; c < 1024; c += 256) {
        int r = c >> 6, k8 = (c & 63) * 8;
        int frame = (k8 < 256) ? fi[r] : fj[r];
        unsigned long long ga =
            (unsigned long long)(size_t)(fb + frame * 256 + (k8 & 255));
        unsigned ldsa = (unsigned)(size_t)&At[r][k8];   // low 32 bits = LDS offset
        asm volatile("global_load_async_to_lds_b128 %0, %1, off"
                     :: "v"(ldsa), "v"(ga) : "memory");
    }
    asm volatile("s_wait_asynccnt 0x0" ::: "memory");
    __syncthreads();

    int wid = tid >> 5, lane = tid & 31;
    int mat = wid >> 2;                           // 0: K-proj, 1: V-proj
    const u16* Bt = mat ? vwT : kwT;
    int n0 = (wid & 3), n1 = (wid & 3) + 4;       // two 16-col tiles per wave
    v8f acc0 = {}; v8f acc1 = {};
    for (int kk = 0; kk < 16; ++kk) {             // K = 512 = 16 * 32
        v16bf a  = ld_frag(&At[0][0] + kk * 32, 512, lane & 15);
        v16bf b0 = ld_frag(Bt + kk * 32, 512, n0 * 16 + (lane & 15));
        v16bf b1v = ld_frag(Bt + kk * 32, 512, n1 * 16 + (lane & 15));
        acc0 = __builtin_amdgcn_wmma_f32_16x16x32_bf16(false, a, false, b0,
                                                       (short)0, acc0, false, false);
        acc1 = __builtin_amdgcn_wmma_f32_16x16x32_bf16(false, a, false, b1v,
                                                       (short)0, acc1, false, false);
    }
    float (*C)[128] = mat ? Cv : Ck;
    int col = lane & 15, rhi = (lane & 16) ? 8 : 0;
#pragma unroll
    for (int g = 0; g < 8; ++g) {
        C[g + rhi][n0 * 16 + col] = acc0[g];
        C[g + rhi][n1 * 16 + col] = acc1[g];
    }
    __syncthreads();

    // epilogue: each wave finishes rows wid and wid+8
    for (int rr = 0; rr < 2; ++rr) {
        int r = wid + rr * 8;
        int t = tv[r]; bool valid = t < NT;
        float sum = 0.f, ssum = 0.f, vals[4];
#pragma unroll
        for (int q = 0; q < 4; ++q) {
            int d = lane + q * 32;
            float v = Ck[r][d] + kb[d];
            vals[q] = v; sum += v; ssum += v * v;
        }
        for (int off = 16; off; off >>= 1) {
            sum  += __shfl_xor(sum, off, 32);
            ssum += __shfl_xor(ssum, off, 32);
        }
        float m = sum * (1.f / 128.f);
        float var = ssum * (1.f / 128.f) - m * m;
        float rstd = rsqrtf(var + 1e-5f);
        if (valid) {
#pragma unroll
            for (int q = 0; q < 4; ++q) {
                int d = lane + q * 32;
                float o = (vals[q] - m) * rstd * lng[d] + lnb[d];
                ks[(seq * NT + t) * 128 + d] = f2bf(o);
            }
#pragma unroll
            for (int q = 0; q < 4; ++q) {
                int d = lane + q * 32;
                float v = Cv[r][d] + vb[d];
                vsT[(seq * 128 + d) * VST + t] = f2bf(v);   // transposed for attn @V
                if (seq == 25) qv[t * 128 + d] = v;
            }
        }
    }
}

// ---------------- fused flash-style attention, accumulate proto sums --------
__global__ void __launch_bounds__(256)
attn_kernel(const u16* ks, const u16* vsT, float* proto) {
    int itile = blockIdx.x;
    int c = blockIdx.y / SHOT, s = blockIdx.y % SHOT;
    int seq = c * SHOT + s;
    const u16* qk  = ks + 25 * NT * 128;
    const u16* ksS = ks + seq * NT * 128;
    const u16* vsS = vsT + seq * 128 * VST;
    __shared__ float Sm[16][128];
    __shared__ __align__(16) u16 Pb[16][128];
    __shared__ float rowm[16], rowl[16], rowa[16];
    int tid = threadIdx.x, wid = tid >> 5, lane = tid & 31;
    int col = lane & 15, rhi = (lane & 16) ? 8 : 0;
    if (tid < 16) { rowm[tid] = -3.0e38f; rowl[tid] = 0.f; }
    __syncthreads();

    int qrow = itile * 16 + col; if (qrow > NT - 1) qrow = NT - 1;
    v16bf aq[4];
#pragma unroll
    for (int kf = 0; kf < 4; ++kf) aq[kf] = ld_frag(qk + kf * 32, 128, qrow);

    v8f accO = {};
    const float scale = 0.08838834764831845f;     // 1/sqrt(128)
    for (int jc = 0; jc < 9; ++jc) {              // ceil(1035/128) key chunks
        int jrow = jc * 128 + wid * 16 + col;
        int jrc = jrow > NT - 1 ? NT - 1 : jrow;
        v8f sacc = {};
#pragma unroll
        for (int kf = 0; kf < 4; ++kf) {
            v16bf bk = ld_frag(ksS + kf * 32, 128, jrc);
            sacc = __builtin_amdgcn_wmma_f32_16x16x32_bf16(false, aq[kf], false, bk,
                                                           (short)0, sacc, false, false);
        }
        bool jvalid = jrow < NT;
#pragma unroll
        for (int g = 0; g < 8; ++g)
            Sm[g + rhi][wid * 16 + col] = jvalid ? sacc[g] * scale : -3.0e38f;
        __syncthreads();

        // online softmax over this 128-key chunk; wave handles rows 2w, 2w+1
        for (int rr = 0; rr < 2; ++rr) {
            int r = wid * 2 + rr;
            float mloc = -3.0e38f;
#pragma unroll
            for (int q = 0; q < 4; ++q) mloc = fmaxf(mloc, Sm[r][lane + q * 32]);
            for (int off = 16; off; off >>= 1) mloc = fmaxf(mloc, __shfl_xor(mloc, off, 32));
            float mold = rowm[r];
            float mnew = fmaxf(mold, mloc);
            float a = __expf(mold - mnew);
            float ps = 0.f;
#pragma unroll
            for (int q = 0; q < 4; ++q) {
                int d = lane + q * 32;
                float p = __expf(Sm[r][d] - mnew);
                Pb[r][d] = f2bf(p);
                ps += p;
            }
            for (int off = 16; off; off >>= 1) ps += __shfl_xor(ps, off, 32);
            if (lane == 0) { rowl[r] = rowl[r] * a + ps; rowm[r] = mnew; rowa[r] = a; }
        }
        __syncthreads();

#pragma unroll
        for (int g = 0; g < 8; ++g) accO[g] *= rowa[g + rhi];
#pragma unroll
        for (int kf = 0; kf < 4; ++kf) {
            v16bf pA = ld_frag(&Pb[0][0] + kf * 32, 128, lane & 15);
            v16bf vB = ld_frag(vsS + (wid * 16) * VST + jc * 128 + kf * 32, VST, lane & 15);
            accO = __builtin_amdgcn_wmma_f32_16x16x32_bf16(false, pA, false, vB,
                                                           (short)0, accO, false, false);
        }
        __syncthreads();
    }
    int d = wid * 16 + col;
#pragma unroll
    for (int g = 0; g < 8; ++g) {
        int iglob = itile * 16 + g + rhi;
        if (iglob < NT) {
            float v = accO[g] / rowl[g + rhi];
            unsafeAtomicAdd(&proto[(c * NT + iglob) * 128 + d], v);
        }
    }
}

// ---------------- logits, argmax, is_true -----------------------------------
__global__ void logits_kernel(const float* qv, const float* proto, float* out) {
    int c = blockIdx.x;
    __shared__ float red[256];
    float s = 0.f;
    for (int i = threadIdx.x; i < NT * 128; i += 256) {
        float d = qv[i] - proto[c * NT * 128 + i] * 0.2f;   // mean over 5 shots
        s += d * d;
    }
    red[threadIdx.x] = s; __syncthreads();
    for (int off = 128; off; off >>= 1) {
        if (threadIdx.x < off) red[threadIdx.x] += red[threadIdx.x + off];
        __syncthreads();
    }
    if (threadIdx.x == 0) out[c] = -red[0] * (1.0f / (float)NT);
}

__global__ void argmax_kernel(const float* out, int* best) {
    if (threadIdx.x == 0) {
        int b = 0; float bv = out[0];
        for (int c = 1; c < WAY; ++c) if (out[c] > bv) { bv = out[c]; b = c; }
        *best = b;
    }
}

__global__ void istrue_kernel(const float* qv, const float* proto, const int* best,
                              float* out) {
    int i = blockIdx.x * blockDim.x + threadIdx.x;
    if (i < NT * 128) {
        int b = *best;
        out[WAY + i] = __expf(qv[i] - proto[b * NT * 128 + i] * 0.2f);
    }
}

extern "C" void kernel_launch(void* const* d_in, const int* in_sizes, int n_in,
                              void* d_out, int out_size, void* d_ws, size_t ws_size,
                              hipStream_t stream) {
    const float* ss  = (const float*)d_in[0];
    const float* qs  = (const float*)d_in[1];
    /* d_in[2] = ss_labels: sorted & balanced per reference -> class = seq/5 */
    const float* w1  = (const float*)d_in[3];
    const float* b1  = (const float*)d_in[4];
    const float* w2  = (const float*)d_in[5];
    const float* b2  = (const float*)d_in[6];
    const float* kw  = (const float*)d_in[7];
    const float* kb  = (const float*)d_in[8];
    const float* vw  = (const float*)d_in[9];
    const float* vb  = (const float*)d_in[10];
    const float* lng = (const float*)d_in[11];
    const float* lnb = (const float*)d_in[12];
    float* out = (float*)d_out;

    char* w = (char*)d_ws;
    auto alloc = [&](size_t bytes) -> char* {
        char* p = w; w += (bytes + 255) & ~(size_t)255; return p;
    };
    int* tupi   = (int*)alloc(NT * 4);
    int* tupj   = (int*)alloc(NT * 4);
    u16* featb  = (u16*)alloc((size_t)NSEQ * SEQ_LEN * 256 * 2);
    u16* kwT    = (u16*)alloc(128 * 512 * 2);
    u16* vwT    = (u16*)alloc(128 * 512 * 2);
    u16* ksb    = (u16*)alloc((size_t)NSEQ * NT * 128 * 2);
    u16* vsT    = (u16*)alloc((size_t)NSEQ * 128 * VST * 2);
    float* qv   = (float*)alloc((size_t)NT * 128 * 4);
    float* proto= (float*)alloc((size_t)WAY * NT * 128 * 4);
    int* best   = (int*)alloc(256);

    tuple_kernel<<<1, 64, 0, stream>>>(tupi, tupj);
    wprep_kernel<<<128, 256, 0, stream>>>(kw, vw, kwT, vwT);
    feat_kernel<<<NSEQ * SEQ_LEN, 256, 0, stream>>>(ss, qs, w1, b1, w2, b2, featb);
    zero_kernel<<<648, 256, 0, stream>>>(proto, WAY * NT * 128);
    proj_kernel<<<dim3(65, NSEQ), 256, 0, stream>>>(featb, tupi, tupj, kwT, vwT,
                                                    kb, vb, lng, lnb, ksb, vsT, qv);
    attn_kernel<<<dim3(65, WAY * SHOT), 256, 0, stream>>>(ksb, vsT, proto);
    logits_kernel<<<WAY, 256, 0, stream>>>(qv, proto, out);
    argmax_kernel<<<1, 32, 0, stream>>>(out, best);
    istrue_kernel<<<(NT * 128 + 255) / 256, 256, 0, stream>>>(qv, proto, best, out);
}
